// MultiHeadAttention_83623013253407
// MI455X (gfx1250) — matrix operand order repaired and checked by reading, not compiled, not run on hardware
//
#include <hip/hip_runtime.h>
#include <stdint.h>

// ---------------- CDNA5 WMMA / TDM plumbing ----------------
typedef __attribute__((ext_vector_type(16))) __bf16 v16bf;
typedef __attribute__((ext_vector_type(8)))  float  v8f;
typedef __attribute__((ext_vector_type(4)))  unsigned int u32x4;
typedef __attribute__((ext_vector_type(8)))  int i32x8;
typedef __attribute__((ext_vector_type(4)))  int i32x4;

union FragU {
  v16bf bf;
  uint4 q[2];
};

static __device__ __forceinline__ unsigned short f2bf(float x) {
  unsigned int u = __builtin_bit_cast(unsigned int, x);
  u += 0x7fffu + ((u >> 16) & 1u);  // round-to-nearest-even
  return (unsigned short)(u >> 16);
}

static __device__ __forceinline__ v8f wmma_bf16(v16bf a, v16bf b, v8f c) {
  return __builtin_amdgcn_wmma_f32_16x16x32_bf16(false, a, false, b, (short)0, c,
                                                 false, false);
}

// Issue one 2D TDM tile load: tile_d1 rows x tile_d0 elems (bf16), row stride
// stride0 elems in global; LDS rows padded +16B (pad_interval=64B, pad=4 dword)
// so the LDS row stride is tile_d0(=32) + 8 = 40 shorts.
static __device__ __forceinline__ void tdm_load_2d(unsigned lds_addr,
                                                   const unsigned short* gptr,
                                                   int tensor_d0, int tensor_d1,
                                                   int stride0, int tile_d0,
                                                   int tile_d1) {
  unsigned long long ga = (unsigned long long)gptr;
  u32x4 g0;
  g0[0] = 1u;                                   // count=1, user descriptor
  g0[1] = lds_addr;                             // LDS byte address
  g0[2] = (unsigned)(ga & 0xffffffffu);         // global_addr[31:0]
  g0[3] = (unsigned)((ga >> 32) & 0x01ffffffu)  // global_addr[56:32]
          | (2u << 30);                         // type = 2 ("image")
  i32x8 g1;
  g1[0] = (1 << 16)      // data_size = 1 -> 2 bytes
        | (1 << 20)      // pad_enable
        | (3 << 22)      // pad_interval: 64B (16 DWORDs)
        | (3 << 25);     // pad_amount: 4 DWORDs (16B)
  g1[1] = (tensor_d0 & 0xffff) << 16;                                // td0 lo
  g1[2] = ((tensor_d0 >> 16) & 0xffff) | ((tensor_d1 & 0xffff) << 16);
  g1[3] = ((tensor_d1 >> 16) & 0xffff) | ((tile_d0 & 0xffff) << 16);
  g1[4] = (tile_d1 & 0xffff);                   // tile_dim1; tile_dim2 = 0
  g1[5] = stride0;                              // tensor_dim0_stride[31:0]
  g1[6] = 0;
  g1[7] = 0;
  i32x4 z4 = {};
#if defined(__clang_major__) && __clang_major__ >= 23
  i32x8 z8 = {};
  __builtin_amdgcn_tensor_load_to_lds(g0, g1, z4, z4, z8, 0);
#else
  __builtin_amdgcn_tensor_load_to_lds(g0, g1, z4, z4, 0);
#endif
}

// ---------------- problem constants ----------------
#define GB   2
#define GS   2048
#define GD   1024
#define GH   16
#define GDK  64
#define GM   (GB * GS)

// ---------------- pack kernels (one-time fp32 -> bf16) ----------------
__global__ __launch_bounds__(256) void cvt_f32_bf16(const float* __restrict__ src,
                                                    unsigned short* __restrict__ dst) {
  const int i = blockIdx.x * 256 + threadIdx.x;  // 8 elements per thread
  const float4* s = (const float4*)src + (size_t)i * 2;
  float4 a = s[0], b = s[1];
  union { uint4 q; unsigned short h[8]; } o;
  o.h[0] = f2bf(a.x); o.h[1] = f2bf(a.y); o.h[2] = f2bf(a.z); o.h[3] = f2bf(a.w);
  o.h[4] = f2bf(b.x); o.h[5] = f2bf(b.y); o.h[6] = f2bf(b.z); o.h[7] = f2bf(b.w);
  ((uint4*)dst)[i] = o.q;
}

// dst[C,R] (bf16) = transpose(src[R,C] (f32))
__global__ __launch_bounds__(256) void transpose_cvt(const float* __restrict__ src,
                                                     unsigned short* __restrict__ dst,
                                                     int R, int C) {
  __shared__ float tile[32][33];
  const int c0 = blockIdx.x * 32, r0 = blockIdx.y * 32;
  const int tx = threadIdx.x;
#pragma unroll
  for (int i = threadIdx.y; i < 32; i += 8)
    tile[i][tx] = src[(size_t)(r0 + i) * C + c0 + tx];
  __syncthreads();
#pragma unroll
  for (int i = threadIdx.y; i < 32; i += 8)
    dst[(size_t)(c0 + i) * R + r0 + tx] = f2bf(tile[tx][i]);
}

// ---------------- TDM-staged 128x128 WMMA GEMM, K-step 32 ----------------
// C[M,N] = A[M,K] * Bt[N,K]^T + bias[N].  A, Bt bf16 (K contiguous).
template <bool OUT_BF16>
__global__ __launch_bounds__(256) void gemm_tdm(const unsigned short* __restrict__ A,
                                                const unsigned short* __restrict__ Bt,
                                                const float* __restrict__ bias,
                                                void* __restrict__ Cp,
                                                int M, int N, int K) {
  constexpr int LS = 40;  // LDS row stride (shorts) = 32 + 8 pad (TDM padding)
  __shared__ unsigned short As[2][128 * LS];
  __shared__ unsigned short Bs[2][128 * LS];

  const int t    = threadIdx.x;
  const int lane = t & 31;
  const int wave = t >> 5;
  const int half = lane >> 4;
  const int ln   = lane & 15;
  const int hi8  = half * 8;
  const int wm   = wave >> 1;
  const int wn   = wave & 1;
  const int m0   = blockIdx.y * 128;
  const int n0   = blockIdx.x * 128;

  v8f acc[2][4];
  v8f vzero = {};
#pragma unroll
  for (int mi = 0; mi < 2; ++mi)
#pragma unroll
    for (int ni = 0; ni < 4; ++ni) acc[mi][ni] = vzero;

  // prologue: fill buffer 0 (wave 0 only; tensor ops are wave-level)
  if (wave == 0) {
    tdm_load_2d((unsigned)(unsigned long long)&As[0][0],
                A + (size_t)m0 * K, K, M, K, 32, 128);
    tdm_load_2d((unsigned)(unsigned long long)&Bs[0][0],
                Bt + (size_t)n0 * K, K, N, K, 32, 128);
  }

  int ib = 0;
  for (int k0 = 0; k0 < K; k0 += 32, ib ^= 1) {
    if (wave == 0) __builtin_amdgcn_s_wait_tensorcnt(0);
    __syncthreads();  // buffer ib ready for everyone; prev reads all done
    if (wave == 0 && k0 + 32 < K) {
      tdm_load_2d((unsigned)(unsigned long long)&As[ib ^ 1][0],
                  A + (size_t)m0 * K + k0 + 32, K, M, K, 32, 128);
      tdm_load_2d((unsigned)(unsigned long long)&Bs[ib ^ 1][0],
                  Bt + (size_t)n0 * K + k0 + 32, K, N, K, 32, 128);
    }

    FragU af[2], bf[4];
#pragma unroll
    for (int mi = 0; mi < 2; ++mi) {
      const unsigned short* p = &As[ib][(wm * 32 + mi * 16 + ln) * LS + hi8];
      af[mi].q[0] = *(const uint4*)p;
      af[mi].q[1] = *(const uint4*)(p + 16);
    }
#pragma unroll
    for (int ni = 0; ni < 4; ++ni) {
      const unsigned short* p = &Bs[ib][(wn * 64 + ni * 16 + ln) * LS + hi8];
      bf[ni].q[0] = *(const uint4*)p;
      bf[ni].q[1] = *(const uint4*)(p + 16);
    }
#pragma unroll
    for (int mi = 0; mi < 2; ++mi)
#pragma unroll
      for (int ni = 0; ni < 4; ++ni)
        acc[mi][ni] = wmma_bf16(af[mi].bf, bf[ni].bf, acc[mi][ni]);
  }

#pragma unroll
  for (int mi = 0; mi < 2; ++mi) {
#pragma unroll
    for (int ni = 0; ni < 4; ++ni) {
      const int col = n0 + wn * 64 + ni * 16 + ln;
      const float bv = bias[col];
#pragma unroll
      for (int j = 0; j < 8; ++j) {
        const int row = m0 + wm * 32 + mi * 16 + hi8 + j;
        const float v = acc[mi][ni][j] + bv;
        if constexpr (OUT_BF16)
          ((unsigned short*)Cp)[(size_t)row * N + col] = f2bf(v);
        else
          ((float*)Cp)[(size_t)row * N + col] = v;
      }
    }
  }
}

// ---------------- V transpose: qkv[.,2048+h*64+d] -> Vt[bh][d][s] ----------------
__global__ __launch_bounds__(256) void transpose_v(const unsigned short* __restrict__ qkv,
                                                   unsigned short* __restrict__ Vt) {
  __shared__ unsigned short tile[32][33];
  const int bh = blockIdx.z;
  const int b = bh >> 4, h = bh & 15;
  const int s0 = blockIdx.x * 32;
  const int d0 = blockIdx.y * 32;
  const int tx = threadIdx.x;
#pragma unroll
  for (int i = threadIdx.y; i < 32; i += 8) {
    const int s = s0 + i;
    tile[i][tx] =
        qkv[(size_t)(b * GS + s) * (3 * GD) + 2 * GD + h * GDK + d0 + tx];
  }
  __syncthreads();
#pragma unroll
  for (int i = threadIdx.y; i < 32; i += 8) {
    const int d = d0 + i;
    Vt[((size_t)bh * GDK + d) * GS + s0 + tx] = tile[tx][i];
  }
}

// ---------------- fused flash-attention (per-wave 16-row q tile) ----------------
__global__ __launch_bounds__(256) void attn_flash(const unsigned short* __restrict__ qkv,
                                                  const int* __restrict__ mask,
                                                  const unsigned short* __restrict__ Vt,
                                                  unsigned short* __restrict__ ctx) {
  __shared__ unsigned short Pb[8 * 16 * 32];

  const int lane = threadIdx.x & 31;
  const int wave = threadIdx.x >> 5;
  const int half = lane >> 4;
  const int ln   = lane & 15;
  const int hi8  = half * 8;
  const int b = blockIdx.z;
  const int h = blockIdx.y;
  const int q0 = blockIdx.x * 128 + wave * 16;

  FragU aq[2];
#pragma unroll
  for (int kd = 0; kd < 2; ++kd) {
    const unsigned short* p =
        qkv + (size_t)(b * GS + q0 + ln) * (3 * GD) + h * GDK + kd * 32 + hi8;
    aq[kd].q[0] = *(const uint4*)p;
    aq[kd].q[1] = *(const uint4*)(p + 16);
  }

  v8f vzero = {};
  v8f acc[4];
#pragma unroll
  for (int tt = 0; tt < 4; ++tt) acc[tt] = vzero;
  float mrow[8], lrow[8];
#pragma unroll
  for (int j = 0; j < 8; ++j) { mrow[j] = -1e30f; lrow[j] = 0.0f; }

  const int* maskbase = mask + (size_t)b * GS * GS;
  unsigned short* Pw = &Pb[wave * 512];
  const unsigned short* vtbase = Vt + (size_t)((b * GH + h) * GDK) * GS;

  for (int kb = 0; kb < GS; kb += 32) {
    v8f s0 = vzero, s1 = vzero;
#pragma unroll
    for (int kd = 0; kd < 2; ++kd) {
      const unsigned short* pk0 = qkv + (size_t)(b * GS + kb + ln) * (3 * GD) +
                                  GD + h * GDK + kd * 32 + hi8;
      FragU bk0, bk1;
      bk0.q[0] = *(const uint4*)pk0;
      bk0.q[1] = *(const uint4*)(pk0 + 16);
      const unsigned short* pk1 = pk0 + (size_t)16 * (3 * GD);
      bk1.q[0] = *(const uint4*)pk1;
      bk1.q[1] = *(const uint4*)(pk1 + 16);
      s0 = wmma_bf16(aq[kd].bf, bk0.bf, s0);
      s1 = wmma_bf16(aq[kd].bf, bk1.bf, s1);
    }

#pragma unroll
    for (int j = 0; j < 8; ++j) {
      const int* mr = maskbase + (size_t)(q0 + hi8 + j) * GS + kb + ln;
      float v0 = (mr[0]  == 0) ? -1e9f : s0[j] * 0.125f;
      float v1 = (mr[16] == 0) ? -1e9f : s1[j] * 0.125f;
      float bm = fmaxf(v0, v1);
      bm = fmaxf(bm, __shfl_xor(bm, 1, 32));
      bm = fmaxf(bm, __shfl_xor(bm, 2, 32));
      bm = fmaxf(bm, __shfl_xor(bm, 4, 32));
      bm = fmaxf(bm, __shfl_xor(bm, 8, 32));
      const float mn = fmaxf(mrow[j], bm);
      const float corr = __expf(mrow[j] - mn);
      const float p0 = __expf(v0 - mn);
      const float p1 = __expf(v1 - mn);
      float rs = p0 + p1;
      rs += __shfl_xor(rs, 1, 32);
      rs += __shfl_xor(rs, 2, 32);
      rs += __shfl_xor(rs, 4, 32);
      rs += __shfl_xor(rs, 8, 32);
      lrow[j] = lrow[j] * corr + rs;
      mrow[j] = mn;
      acc[0][j] *= corr; acc[1][j] *= corr;
      acc[2][j] *= corr; acc[3][j] *= corr;
      Pw[(hi8 + j) * 32 + ln]      = f2bf(p0);
      Pw[(hi8 + j) * 32 + 16 + ln] = f2bf(p1);
    }

    FragU pa;
    const unsigned short* pp = &Pw[ln * 32 + hi8];
    pa.q[0] = *(const uint4*)pp;
    pa.q[1] = *(const uint4*)(pp + 16);

#pragma unroll
    for (int tt = 0; tt < 4; ++tt) {
      const unsigned short* pv =
          vtbase + (size_t)(tt * 16 + ln) * GS + kb + hi8;
      FragU bv;
      bv.q[0] = *(const uint4*)pv;
      bv.q[1] = *(const uint4*)(pv + 16);
      acc[tt] = wmma_bf16(pa.bf, bv.bf, acc[tt]);
    }
  }

#pragma unroll
  for (int tt = 0; tt < 4; ++tt) {
#pragma unroll
    for (int j = 0; j < 8; ++j) {
      const float v = acc[tt][j] / lrow[j];
      ctx[(size_t)(b * GS + q0 + hi8 + j) * GD + h * GDK + tt * 16 + ln] =
          f2bf(v);
    }
  }
}

// ---------------- host launch ----------------
extern "C" void kernel_launch(void* const* d_in, const int* in_sizes, int n_in,
                              void* d_out, int out_size, void* d_ws, size_t ws_size,
                              hipStream_t stream) {
  (void)in_sizes; (void)n_in; (void)out_size; (void)ws_size;
  const float* query = (const float*)d_in[0];
  const int*   mask  = (const int*)d_in[1];
  const float* Wqkv  = (const float*)d_in[2];
  const float* bqkv  = (const float*)d_in[3];
  const float* Wout  = (const float*)d_in[4];
  const float* bout  = (const float*)d_in[5];
  float* out = (float*)d_out;

  // ws: qkv[4096,3072] | Vt[32,64,2048] | ctx[4096,1024] | qb[4096,1024]
  //     | Wt1[3072,1024] | Wt2[1024,1024]   (all bf16)
  char* ws = (char*)d_ws;
  unsigned short* qkv = (unsigned short*)ws;                       // 25165824 B
  unsigned short* Vt  = (unsigned short*)(ws + 25165824);          //  8388608 B
  unsigned short* ctx = (unsigned short*)(ws + 33554432);          //  8388608 B
  unsigned short* qb  = (unsigned short*)(ws + 41943040);          //  8388608 B
  unsigned short* Wt1 = (unsigned short*)(ws + 50331648);          //  6291456 B
  unsigned short* Wt2 = (unsigned short*)(ws + 56623104);          //  2097152 B

  // 0) one-time packs: query -> bf16; W_qkv, W_out -> transposed bf16
  cvt_f32_bf16<<<dim3((GM * GD) / (8 * 256)), 256, 0, stream>>>(query, qb);
  transpose_cvt<<<dim3(3 * GD / 32, GD / 32), dim3(32, 8), 0, stream>>>(
      Wqkv, Wt1, GD, 3 * GD);
  transpose_cvt<<<dim3(GD / 32, GD / 32), dim3(32, 8), 0, stream>>>(
      Wout, Wt2, GD, GD);

  // 1) qkv = query @ W_qkv + b_qkv (TDM-staged WMMA GEMM, bf16 out)
  gemm_tdm<true><<<dim3(3 * GD / 128, GM / 128), 256, 0, stream>>>(
      qb, Wt1, bqkv, qkv, GM, 3 * GD, GD);

  // 2) Vt[bh][d][s]
  transpose_v<<<dim3(GS / 32, GDK / 32, GB * GH), dim3(32, 8), 0, stream>>>(qkv, Vt);

  // 3) fused flash attention -> ctx
  attn_flash<<<dim3(GS / 128, GH, GB), 256, 0, stream>>>(qkv, mask, Vt, ctx);

  // 4) out = ctx @ W_out + b_out (TDM-staged WMMA GEMM, f32 out)
  gemm_tdm<false><<<dim3(GD / 128, GM / 128), 256, 0, stream>>>(
      ctx, Wt2, bout, out, GM, GD, GD);
}
